// VectorQuantizer_31653908972291
// MI455X (gfx1250) — matrix operand order repaired and checked by reading, not compiled, not run on hardware
//
#include <hip/hip_runtime.h>

// ---------------------------------------------------------------------------
// VQ-VAE vector quantizer for MI455X (gfx1250, wave32, WMMA).
//
// Distances: d(r,n) = ||e_n||^2 - 2 * <z_r, e_n>   (||z_r||^2 constant per row)
// GEMM 16384x8192x256 done with V_WMMA_F32_16X16X32_BF16 using split-bf16
// (hi+lo) operands for ~fp32-accurate dot products:
//   z.e ~= zh.eh + zl.eh + zh.el     (zl.el term ~2^-32, dropped)
// ---------------------------------------------------------------------------

typedef __attribute__((ext_vector_type(16))) __bf16 v16bf;
typedef __attribute__((ext_vector_type(8)))  float  v8f;
typedef unsigned short u16;
typedef unsigned int   u32;

#define Bsz 16
#define Csz 256
#define Tsz 1024
#define Nsz 8192
#define Msz (Bsz * Tsz)          // 16384 rows
#define NELEM (Msz * Csz)        // 4194304

__device__ __forceinline__ u16 bf16_rne(float f) {
    u32 u = __float_as_uint(f);
    return (u16)((u + 0x7FFFu + ((u >> 16) & 1u)) >> 16);
}

// ---- Stage 1: codebook split + row norms ----------------------------------
__global__ void vq_prep_emb(const float* __restrict__ emb, float* __restrict__ enorm,
                            u16* __restrict__ embh, u16* __restrict__ embl) {
    int n = blockIdx.x, c = threadIdx.x;
    float e  = emb[n * Csz + c];
    u16  hi  = bf16_rne(e);
    float ef = __uint_as_float(((u32)hi) << 16);
    u16  lo  = bf16_rne(e - ef);
    embh[n * Csz + c] = hi;
    embl[n * Csz + c] = lo;
    __shared__ float red[256];
    red[c] = e * e;
    __syncthreads();
    for (int s = 128; s > 0; s >>= 1) { if (c < s) red[c] += red[c + s]; __syncthreads(); }
    if (c == 0) enorm[n] = red[0];
}

// ---- Stage 2: transpose z [B,C,T] -> [B*T, C] with hi/lo split ------------
__global__ void vq_prep_z(const float* __restrict__ z, u16* __restrict__ zh,
                          u16* __restrict__ zl) {
    __shared__ float tile[32][33];
    int blk = blockIdx.x;
    int b   = blk >> 8;              // (C/32)*(T/32) = 8*32 = 256 tiles per batch
    int rem = blk & 255;
    int c0  = (rem >> 5) << 5;
    int t0  = (rem & 31) << 5;
    int tid = threadIdx.x;
#pragma unroll
    for (int j = 0; j < 4; j++) {
        int id = tid + j * 256;
        int cl = id >> 5, tl = id & 31;     // read: contiguous t across lanes
        tile[cl][tl] = z[((size_t)b * Csz + c0 + cl) * Tsz + t0 + tl];
    }
    __syncthreads();
#pragma unroll
    for (int j = 0; j < 4; j++) {
        int id = tid + j * 256;
        int tl = id >> 5, cl = id & 31;     // write: contiguous c across lanes
        float e  = tile[cl][tl];
        u16  hi  = bf16_rne(e);
        float ef = __uint_as_float(((u32)hi) << 16);
        u16  lo  = bf16_rne(e - ef);
        size_t o = ((size_t)(b * Tsz + t0 + tl)) * Csz + c0 + cl;
        zh[o] = hi; zl[o] = lo;
    }
}

// ---- Stage 3: WMMA distance search + argmin -------------------------------
// 1 block = one 16-row M tile; 4 waves split the 8192 codes (2048 each).
__global__ void __launch_bounds__(128) vq_search(
    const u16* __restrict__ zh, const u16* __restrict__ zl,
    const u16* __restrict__ embh, const u16* __restrict__ embl,
    const float* __restrict__ enorm, int* __restrict__ idx_out) {
    __shared__ u16 Ah[16 * Csz];
    __shared__ u16 Al[16 * Csz];
    __shared__ float sdist[4][16];
    __shared__ int   sidx[4][16];

    const int m0  = blockIdx.x * 16;
    const int tid = threadIdx.x;

    // Stage the 16x256 A tile (hi+lo) into LDS: 8KB each, b128 copies.
    {
        const uint4* gh = (const uint4*)(zh + (size_t)m0 * Csz);
        const uint4* gl = (const uint4*)(zl + (size_t)m0 * Csz);
        uint4* sh = (uint4*)Ah;
        uint4* sl = (uint4*)Al;
#pragma unroll
        for (int j = 0; j < 4; j++) {
            sh[tid + j * 128] = gh[tid + j * 128];
            sl[tid + j * 128] = gl[tid + j * 128];
        }
    }
    __syncthreads();

    const int lane = tid & 31;
    const int wave = tid >> 5;
    const int half = lane >> 4;      // lanes 16-31 hold the upper K group
    const int mrow = lane & 15;      // A-matrix row for this lane

    // Build A fragments in registers (contiguous K order, matching B):
    //   VGPR v : K = 2v + half*16
    union AF { v16bf v; u32 u[8]; };
    AF ah[8], al[8];
#pragma unroll
    for (int s = 0; s < 8; s++) {
#pragma unroll
        for (int v = 0; v < 8; v++) {
            int kb = s * 32 + 2 * v + half * 16;
            ah[s].u[v] = *(const u32*)(&Ah[mrow * Csz + kb]);
            al[s].u[v] = *(const u32*)(&Al[mrow * Csz + kb]);
        }
    }

    float best[8];
    int   bidx[8];
#pragma unroll
    for (int i = 0; i < 8; i++) { best[i] = 3.4e38f; bidx[i] = 0x7fffffff; }

    const int nlocal = lane & 15;    // B/C/D column for this lane
    for (int t = 0; t < 128; t++) {  // 128 N-tiles of 16 codes per wave
        const int n = wave * 2048 + t * 16 + nlocal;
        v8f acc = {};
#pragma unroll
        for (int s = 0; s < 8; s++) {
            // B fragment: lane holds column n, elements e -> K = s*32 + half*16 + e
            union BF { v16bf v; uint4 q[2]; } bh, bl;
            const uint4* ph = (const uint4*)(embh + (size_t)n * Csz + s * 32 + half * 16);
            const uint4* pl = (const uint4*)(embl + (size_t)n * Csz + s * 32 + half * 16);
            bh.q[0] = ph[0]; bh.q[1] = ph[1];
            bl.q[0] = pl[0]; bl.q[1] = pl[1];
            acc = __builtin_amdgcn_wmma_f32_16x16x32_bf16(false, ah[s].v, false, bh.v,
                                                          (short)0, acc, false, false);
            acc = __builtin_amdgcn_wmma_f32_16x16x32_bf16(false, al[s].v, false, bh.v,
                                                          (short)0, acc, false, false);
            acc = __builtin_amdgcn_wmma_f32_16x16x32_bf16(false, ah[s].v, false, bl.v,
                                                          (short)0, acc, false, false);
        }
        const float en = enorm[n];
#pragma unroll
        for (int i = 0; i < 8; i++) {       // acc[i] -> row m0 + i + 8*half, col n
            float d = en - 2.0f * acc[i];
            if (d < best[i]) { best[i] = d; bidx[i] = n; }   // first-min on ties
        }
    }

    // Reduce across the 16 lanes sharing the same rows (same half).
#pragma unroll
    for (int mask = 1; mask < 16; mask <<= 1) {
#pragma unroll
        for (int i = 0; i < 8; i++) {
            float od = __shfl_xor(best[i], mask, 32);
            int   oi = __shfl_xor(bidx[i], mask, 32);
            if (od < best[i] || (od == best[i] && oi < bidx[i])) { best[i] = od; bidx[i] = oi; }
        }
    }
    if (nlocal == 0) {
#pragma unroll
        for (int i = 0; i < 8; i++) {
            sdist[wave][i + 8 * half] = best[i];
            sidx [wave][i + 8 * half] = bidx[i];
        }
    }
    __syncthreads();
    if (tid < 16) {
        float bd = sdist[0][tid];
        int   bi = sidx [0][tid];
#pragma unroll
        for (int w = 1; w < 4; w++) {
            float od = sdist[w][tid];
            int   oi = sidx [w][tid];
            if (od < bd || (od == bd && oi < bi)) { bd = od; bi = oi; }
        }
        idx_out[m0 + tid] = bi;
    }
}

// ---- Stage 4: gather z_q, compute loss ------------------------------------
__global__ void vq_zero(float* loss) { *loss = 0.0f; }

__global__ void vq_gather(const float* __restrict__ z, const float* __restrict__ emb,
                          const int* __restrict__ idxv, float* __restrict__ out,
                          float* __restrict__ loss) {
    int b = blockIdx.x >> 2;                       // 4 blocks of 256 t per batch
    int t = (blockIdx.x & 3) * 256 + threadIdx.x;
    int r = b * Tsz + t;
    int id = idxv[r];
    const float* e = emb + (size_t)id * Csz;
    float sq = 0.0f;
    for (int c = 0; c < Csz; c++) {
        float q  = e[c];
        size_t o = ((size_t)b * Csz + c) * Tsz + t;   // coalesced over t
        float zv = z[o];
        out[o] = q;                                    // straight-through value == emb[idx]
        float d = q - zv;
        sq += d * d;
    }
    __shared__ float red[256];
    red[threadIdx.x] = sq;
    __syncthreads();
    for (int s = 128; s > 0; s >>= 1) {
        if (threadIdx.x < s) red[threadIdx.x] += red[threadIdx.x + s];
        __syncthreads();
    }
    // loss = (0.25 + 1.0) * mean(diff^2)
    if (threadIdx.x == 0) atomicAdd(loss, red[0] * (1.25f / (float)NELEM));
}

// ---------------------------------------------------------------------------
extern "C" void kernel_launch(void* const* d_in, const int* in_sizes, int n_in,
                              void* d_out, int out_size, void* d_ws, size_t ws_size,
                              hipStream_t stream) {
    const float* z   = (const float*)d_in[0];   // [16, 256, 1024]
    const float* emb = (const float*)d_in[1];   // [8192, 256]

    float* out  = (float*)d_out;                // z_q: NELEM floats
    float* loss = out + NELEM;                  // 1 float
    int*   idxv = (int*)(out + NELEM + 1);      // 16384 int32

    // Workspace layout (~24 MB)
    char* ws = (char*)d_ws;
    float* enorm = (float*)ws;                                   // 32 KB
    u16* embh = (u16*)(ws + 32768);                              // 4 MB
    u16* embl = (u16*)(ws + 32768 + (size_t)Nsz * Csz * 2);      // 4 MB
    u16* zh   = (u16*)(ws + 32768 + (size_t)Nsz * Csz * 4);      // 8 MB
    u16* zl   = (u16*)(ws + 32768 + (size_t)Nsz * Csz * 4 + (size_t)Msz * Csz * 2);

    vq_prep_emb<<<Nsz, 256, 0, stream>>>(emb, enorm, embh, embl);
    vq_prep_z<<<Bsz * (Csz / 32) * (Tsz / 32), 256, 0, stream>>>(z, zh, zl);
    vq_search<<<Msz / 16, 128, 0, stream>>>(zh, zl, embh, embl, enorm, idxv);
    vq_zero<<<1, 1, 0, stream>>>(loss);
    vq_gather<<<Bsz * (Tsz / 256), 256, 0, stream>>>(z, emb, idxv, out, loss);
}
